// PairStack_11441792877253
// MI455X (gfx1250) — matrix operand order repaired and checked by reading, not compiled, not run on hardware
//
#include <hip/hip_runtime.h>

// ---------------------------------------------------------------------------
// CDNA5 (gfx1250) PairStack: bf16 WMMA GEMMs + fp32 epilogues. Wave32.
// All matmul work through v_wmma_f32_16x16x32_bf16; tile loads are
// unconditional 128-bit vector loads matching the ISA A/B VGPR layouts.
// ---------------------------------------------------------------------------

typedef __bf16 bf16_t;
typedef bf16_t v16bf __attribute__((ext_vector_type(16)));
typedef float  v8f   __attribute__((ext_vector_type(8)));
typedef unsigned short u16;
typedef u16 u16x8 __attribute__((ext_vector_type(8)));

struct P16 { u16x8 lo, hi; };

#define NRES 256
#define CZ   128
#define CS   384
#define NN   (NRES * NRES)
#define NHEAD 4
#define CATT  32
#define INF_F 1e9f

// ---------------- scalar bf16 helpers (no __bf16 arithmetic) ---------------
__device__ __forceinline__ float bf2f(u16 u) {
  unsigned v = ((unsigned)u) << 16;
  return __builtin_bit_cast(float, v);
}
__device__ __forceinline__ u16 f2bfu(float f) {
  unsigned u = __builtin_bit_cast(unsigned, f);
  unsigned r = (u + 0x7fffu + ((u >> 16) & 1u)) >> 16;
  return (u16)r;
}
__device__ __forceinline__ bf16_t bcb(u16 u) { return __builtin_bit_cast(bf16_t, u); }

__device__ __forceinline__ v8f zero8() {
  v8f z;
#pragma unroll
  for (int i = 0; i < 8; ++i) z[i] = 0.0f;
  return z;
}

__device__ __forceinline__ v8f wmma_bf16(v16bf a, v16bf b, v8f c) {
  return __builtin_amdgcn_wmma_f32_16x16x32_bf16(false, a, false, b, (short)0, c,
                                                 false, false);
}

// A tile 16x32 bf16, contiguous K (kstride==1): lane m = lane&15, half = lane>>4,
// elems 0..7 -> K=half*8+e, elems 8..15 -> K=16+half*8+(e-8). Two 16B runs.
__device__ __forceinline__ v16bf ldA_c(const u16* base, size_t mstride) {
  int lane = threadIdx.x & 31;
  const u16* p = base + (size_t)(lane & 15) * mstride + (size_t)(lane >> 4) * 8;
  u16x8 lo = *(const u16x8*)p;
  u16x8 hi = *(const u16x8*)(p + 16);
  return __builtin_bit_cast(v16bf, P16{lo, hi});
}

// B tile 32x16 bf16, contiguous K: lane n = lane&15, half = lane>>4,
// elems 0..7 -> K=half*8+e, elems 8..15 -> K=16+half*8+(e-8). Two 16B runs.
// element (k,n) = base[n*nstride + k].
__device__ __forceinline__ v16bf ldB_c(const u16* base, size_t nstride) {
  int lane = threadIdx.x & 31;
  const u16* p = base + (size_t)(lane & 15) * nstride + (size_t)(lane >> 4) * 8;
  u16x8 lo = *(const u16x8*)p;
  u16x8 hi = *(const u16x8*)(p + 16);
  return __builtin_bit_cast(v16bf, P16{lo, hi});
}

// B tile with strided K (element (k,n) = base[n*nstride + k*kstride]); scalar.
__device__ __forceinline__ v16bf ldB_s(const u16* base, size_t nstride, size_t kstride) {
  int lane = threadIdx.x & 31;
  int n = lane & 15, half = lane >> 4;
  const u16* p = base + (size_t)n * nstride;
  v16bf r;
#pragma unroll
  for (int e = 0; e < 16; ++e) r[e] = bcb(p[(size_t)((e < 8) ? (half * 8 + e) : (16 + half * 8 + e - 8)) * kstride]);
  return r;
}

// ---------------------------------------------------------------------------
// weight convert + transpose + pad: f32 W[K,N] -> bf16 Wt[Npad,K] (rows >= N zero)
// ---------------------------------------------------------------------------
__global__ void k_cvt_wt(const float* __restrict__ w, u16* __restrict__ wt, int K,
                         int Nn, int Npad) {
  int idx = blockIdx.x * 256 + threadIdx.x;
  if (idx >= K * Npad) return;
  int n = idx / K, k = idx % K;
  wt[idx] = (n < Nn) ? f2bfu(w[(size_t)k * Nn + n]) : (u16)0;
}

// ---------------------------------------------------------------------------
// LayerNorm over last dim C, fp32 in -> bf16 out. One wave per row.
// ---------------------------------------------------------------------------
__global__ void k_ln(const float* __restrict__ x, const float* __restrict__ g,
                     const float* __restrict__ b, u16* __restrict__ out, int C, int rows) {
  int wave = threadIdx.x >> 5, lane = threadIdx.x & 31;
  int r = blockIdx.x * (blockDim.x >> 5) + wave;
  if (r >= rows) return;
  const float* xr = x + (size_t)r * C;
  float s = 0.f, s2 = 0.f;
  for (int c = lane; c < C; c += 32) { float v = xr[c]; s += v; s2 += v * v; }
#pragma unroll
  for (int o = 16; o; o >>= 1) { s += __shfl_xor(s, o, 32); s2 += __shfl_xor(s2, o, 32); }
  float mu = s / (float)C;
  float var = s2 / (float)C - mu * mu;
  float inv = rsqrtf(var + 1e-5f);
  u16* orow = out + (size_t)r * C;
  for (int c = lane; c < C; c += 32)
    orow[c] = f2bfu((xr[c] - mu) * inv * g[c] + b[c]);
}

// ---------------------------------------------------------------------------
// Generic GEMM: out = epilogue(A[M,K]bf16 @ Wt[Npad,K]bf16^T).
// One wave -> one 16x16 tile. blockDim = 32*NT, grid.x = M/16.
// outmode: 0 store bf16, 1 store f32, 2 += f32, 3 += f32 at transposed row.
// ---------------------------------------------------------------------------
__global__ void k_gemm(const u16* __restrict__ A, const u16* __restrict__ Wt, int K,
                       int Nout, const float* __restrict__ bias, float scale, int act,
                       const float* __restrict__ maskrow, int outmode,
                       float* __restrict__ outf, u16* __restrict__ outh) {
  int nt = threadIdx.x >> 5;
  int m0 = blockIdx.x * 16;
  int n0 = nt * 16;
  v8f acc = zero8();
  for (int k0 = 0; k0 < K; k0 += 32) {
    v16bf a = ldA_c(A + (size_t)m0 * K + k0, K);
    v16bf b = ldB_c(Wt + (size_t)n0 * K + k0, K);
    acc = wmma_bf16(a, b, acc);
  }
  int lane = threadIdx.x & 31;
  int n = lane & 15, mh = (lane >> 4) * 8;
#pragma unroll
  for (int r = 0; r < 8; ++r) {
    int m = m0 + r + mh;
    int nn = n0 + n;
    if (nn >= Nout) continue;
    float v = acc[r] * scale + (bias ? bias[nn] : 0.f);
    if (act == 1) v = 1.f / (1.f + __expf(-v));
    else if (act == 2) v = fmaxf(v, 0.f);
    if (maskrow) v *= maskrow[m];
    if (outmode == 0) outh[(size_t)m * Nout + nn] = f2bfu(v);
    else if (outmode == 1) outf[(size_t)m * Nout + nn] = v;
    else if (outmode == 2) outf[(size_t)m * Nout + nn] += v;
    else {
      size_t tm = (size_t)(m & (NRES - 1)) * NRES + (m >> 8);
      outf[tm * Nout + nn] += v;
    }
  }
}

// ---------------------------------------------------------------------------
// Dual GEMM: P = A1@W1+b1, G = A2@W2+b2, out = sigmoid(G)*P [*mask]
// mode 0: store bf16, mode 1: += f32. Nout = 128, blockDim 256.
// ---------------------------------------------------------------------------
__global__ void k_gemm_dual(const u16* __restrict__ A1, const u16* __restrict__ W1t,
                            const float* __restrict__ b1, const u16* __restrict__ A2,
                            const u16* __restrict__ W2t, const float* __restrict__ b2,
                            int K, const float* __restrict__ maskrow, int mode,
                            float* __restrict__ outf, u16* __restrict__ outh) {
  int nt = threadIdx.x >> 5;
  int m0 = blockIdx.x * 16;
  int n0 = nt * 16;
  v8f p = zero8(), g = zero8();
  for (int k0 = 0; k0 < K; k0 += 32) {
    v16bf a1 = ldA_c(A1 + (size_t)m0 * K + k0, K);
    v16bf w1 = ldB_c(W1t + (size_t)n0 * K + k0, K);
    p = wmma_bf16(a1, w1, p);
    v16bf a2 = ldA_c(A2 + (size_t)m0 * K + k0, K);
    v16bf w2 = ldB_c(W2t + (size_t)n0 * K + k0, K);
    g = wmma_bf16(a2, w2, g);
  }
  int lane = threadIdx.x & 31;
  int n = lane & 15, mh = (lane >> 4) * 8;
#pragma unroll
  for (int r = 0; r < 8; ++r) {
    int m = m0 + r + mh;
    int nn = n0 + n;
    float pv = p[r] + b1[nn];
    float gv = g[r] + b2[nn];
    gv = 1.f / (1.f + __expf(-gv));
    float v = gv * pv;
    if (maskrow) v *= maskrow[m];
    if (mode == 0) outh[(size_t)m * CZ + nn] = f2bfu(v);
    else outf[(size_t)m * CZ + nn] += v;
  }
}

// ---------------------------------------------------------------------------
// OuterProduct GEMM: row r=(i*256+j): A[r,k] = a[i,k]*b[j,k]; z += A @ Wo + bo
// ---------------------------------------------------------------------------
__global__ void k_outer(const u16* __restrict__ as, const u16* __restrict__ bs,
                        const u16* __restrict__ Wot, const float* __restrict__ bo,
                        float* __restrict__ z) {
  int nt = threadIdx.x >> 5;
  int n0 = nt * 16;
  int m0 = blockIdx.x * 16;
  int lane = threadIdx.x & 31;
  int m = m0 + (lane & 15);
  int half = lane >> 4;
  int i = m >> 8, j = m & (NRES - 1);
  v8f acc = zero8();
  for (int k0 = 0; k0 < CZ; k0 += 32) {
    const u16* ar = as + (size_t)i * CZ + k0 + half * 8;
    const u16* br = bs + (size_t)j * CZ + k0 + half * 8;
    u16x8 alo = *(const u16x8*)ar, ahi = *(const u16x8*)(ar + 16);
    u16x8 blo = *(const u16x8*)br, bhi = *(const u16x8*)(br + 16);
    v16bf a;
#pragma unroll
    for (int e = 0; e < 8; ++e) {
      a[e] = bcb(f2bfu(bf2f(alo[e]) * bf2f(blo[e])));
      a[e + 8] = bcb(f2bfu(bf2f(ahi[e]) * bf2f(bhi[e])));
    }
    v16bf b = ldB_c(Wot + (size_t)n0 * CZ + k0, CZ);
    acc = wmma_bf16(a, b, acc);
  }
  int n = lane & 15, mh = (lane >> 4) * 8;
#pragma unroll
  for (int r = 0; r < 8; ++r) {
    int mm = m0 + r + mh;
    z[(size_t)mm * CZ + n0 + n] += acc[r] + bo[n0 + n];
  }
}

// ---------------------------------------------------------------------------
// Triangle einsum, LDS-staged: outgoing t[i,j,c] = sum_k a[i,k,c] b[j,k,c]
//                              incoming t[i,j,c] = sum_k a[k,i,c] b[k,j,c]
// grid (jt=16, it=16, cg=16), 8 waves/block (one channel each).
// Stage [16 rows x 32 k x 8 ch] tiles with contiguous b128 loads (ch innermost),
// transpose to channel-major in LDS, feed WMMA with vector ds loads.
// ---------------------------------------------------------------------------
__global__ void k_triein(const u16* __restrict__ ab, const u16* __restrict__ bb,
                         float* __restrict__ t, int incoming) {
  __shared__ __align__(16) u16 As[8 * 16 * 32];  // [c][m*32+k]
  __shared__ __align__(16) u16 Bs[8 * 16 * 32];  // [c][n*32+k]
  int w = threadIdx.x >> 5, lane = threadIdx.x & 31;
  int c0 = blockIdx.z * 8;
  int i0 = blockIdx.y * 16, j0 = blockIdx.x * 16;
  v8f acc = zero8();
  for (int kt = 0; kt < 8; ++kt) {
    int k0 = kt * 32;
    __syncthreads();
    for (int idx = threadIdx.x; idx < 512; idx += 256) {
      int m = idx >> 5, kk = idx & 31;
      size_t ga = !incoming ? (((size_t)(i0 + m) * NRES + (k0 + kk)) * CZ + c0)
                            : (((size_t)(k0 + kk) * NRES + (i0 + m)) * CZ + c0);
      size_t gb = !incoming ? (((size_t)(j0 + m) * NRES + (k0 + kk)) * CZ + c0)
                            : (((size_t)(k0 + kk) * NRES + (j0 + m)) * CZ + c0);
      u16x8 va = *(const u16x8*)(ab + ga);
      u16x8 vb = *(const u16x8*)(bb + gb);
#pragma unroll
      for (int cc = 0; cc < 8; ++cc) {
        As[cc * 512 + idx] = va[cc];
        Bs[cc * 512 + idx] = vb[cc];
      }
    }
    __syncthreads();
    v16bf a = ldA_c(As + w * 512, 32);
    v16bf b = ldB_c(Bs + w * 512, 32);
    acc = wmma_bf16(a, b, acc);
  }
  int c = c0 + w;
  int n = lane & 15, mh = (lane >> 4) * 8;
#pragma unroll
  for (int r = 0; r < 8; ++r) {
    int ii = i0 + r + mh, jj = j0 + n;
    t[((size_t)ii * NRES + jj) * CZ + c] = acc[r];
  }
}

// ---------------------------------------------------------------------------
// Triangle attention core: one wave per (j-tile, head, i).
// scores (WMMA) -> LDS -> +tri_bias +mask_bias -> softmax -> probs bf16 (LDS)
// -> o = P @ V (WMMA) -> gate -> obuf bf16.
// transposed=1: operate on z^T coordinates (row(j) stride = N*CZ).
// ---------------------------------------------------------------------------
__global__ void k_att(const u16* __restrict__ qb, const u16* __restrict__ kb,
                      const u16* __restrict__ vb, const u16* __restrict__ gb,
                      const float* __restrict__ tb, const float* __restrict__ mask,
                      u16* __restrict__ ob, int transposed) {
  __shared__ __align__(16) float sc[16 * NRES];
  __shared__ __align__(16) u16 pr[16 * NRES];
  int jt = blockIdx.x, h = blockIdx.y, i = blockIdx.z;
  int lane = threadIdx.x & 31;
  int j0 = jt * 16;
  size_t rstride = transposed ? (size_t)NRES * CZ : (size_t)CZ;
  size_t boff = transposed ? (size_t)i * CZ : (size_t)i * NRES * CZ;
  const u16* qrow = qb + boff + (size_t)j0 * rstride + h * CATT;

  // scores: 16 key-tiles of 16, K = CATT = 32 (single WMMA each)
  for (int kt = 0; kt < 16; ++kt) {
    v16bf a = ldA_c(qrow, rstride);
    v16bf b = ldB_c(kb + boff + (size_t)(kt * 16) * rstride + h * CATT, rstride);
    v8f acc = zero8();
    acc = wmma_bf16(a, b, acc);
    int n = lane & 15, mh = (lane >> 4) * 8;
#pragma unroll
    for (int r = 0; r < 8; ++r) sc[(r + mh) * NRES + kt * 16 + n] = acc[r];
  }

  // bias: tri-bias (j,k,h) + mask bias (i,k)
  for (int e = lane; e < 16 * NRES; e += 32) {
    int jj = e >> 8, k = e & (NRES - 1);
    int jg = j0 + jj;
    size_t ti = transposed ? ((size_t)k * NRES + jg) : ((size_t)jg * NRES + k);
    sc[e] += tb[ti * NHEAD + h] + INF_F * (mask[(size_t)i * NRES + k] - 1.f);
  }

  // softmax per row, write probs as bf16
  for (int jj = 0; jj < 16; ++jj) {
    float* row = sc + jj * NRES;
    float ev[8];
    float mx = -3.4e38f;
#pragma unroll
    for (int t = 0; t < 8; ++t) { ev[t] = row[lane + t * 32]; mx = fmaxf(mx, ev[t]); }
#pragma unroll
    for (int o = 16; o; o >>= 1) mx = fmaxf(mx, __shfl_xor(mx, o, 32));
    float s = 0.f;
#pragma unroll
    for (int t = 0; t < 8; ++t) { ev[t] = __expf(ev[t] - mx); s += ev[t]; }
#pragma unroll
    for (int o = 16; o; o >>= 1) s += __shfl_xor(s, o, 32);
    float inv = 1.f / s;
#pragma unroll
    for (int t = 0; t < 8; ++t) pr[jj * NRES + lane + t * 32] = f2bfu(ev[t] * inv);
  }

  // o = probs @ V  (K=256 in 8 steps), gate with sigmoid(g), store bf16
  for (int ntc = 0; ntc < 2; ++ntc) {
    v8f acc = zero8();
    for (int kt = 0; kt < 8; ++kt) {
      v16bf a = ldA_c(pr + kt * 32, NRES);
      // V element (k, c) = vb[boff + (kt*32+k)*rstride + h*CATT + ntc*16 + c]
      v16bf b = ldB_s(vb + boff + (size_t)(kt * 32) * rstride + h * CATT + ntc * 16,
                      1, rstride);
      acc = wmma_bf16(a, b, acc);
    }
    int n = lane & 15, mh = (lane >> 4) * 8;
#pragma unroll
    for (int r = 0; r < 8; ++r) {
      int jj = r + mh;
      int jg = j0 + jj;
      int cc = h * CATT + ntc * 16 + n;
      size_t grow = transposed ? ((size_t)jg * NRES + i) : ((size_t)i * NRES + jg);
      float gv = bf2f(gb[grow * CZ + cc]);
      ob[((size_t)i * NRES + jg) * CZ + cc] = f2bfu(acc[r] * gv);
    }
  }
}

// ---------------------------------------------------------------------------
// Pair transition, fused: h = relu(x@W1+b1) (16x512 in LDS), z += (h@W2+b2)*mask
// block = 256 threads (8 waves) per 16-row tile.
// ---------------------------------------------------------------------------
__global__ void k_pt(const u16* __restrict__ X, const u16* __restrict__ W1t,
                     const float* __restrict__ b1, const u16* __restrict__ W2t,
                     const float* __restrict__ b2, const float* __restrict__ maskrow,
                     float* __restrict__ z) {
  __shared__ __align__(16) u16 h1[16 * 512];
  int m0 = blockIdx.x * 16;
  int w = threadIdx.x >> 5, lane = threadIdx.x & 31;
  int n = lane & 15, mh = (lane >> 4) * 8;

  for (int t = w; t < 32; t += 8) {
    int n0 = t * 16;
    v8f acc = zero8();
    for (int k0 = 0; k0 < CZ; k0 += 32) {
      v16bf a = ldA_c(X + (size_t)m0 * CZ + k0, CZ);
      v16bf b = ldB_c(W1t + (size_t)n0 * CZ + k0, CZ);
      acc = wmma_bf16(a, b, acc);
    }
#pragma unroll
    for (int r = 0; r < 8; ++r) {
      float v = fmaxf(acc[r] + b1[n0 + n], 0.f);
      h1[(r + mh) * 512 + n0 + n] = f2bfu(v);
    }
  }
  __syncthreads();
  {
    int n0 = w * 16;
    v8f acc = zero8();
    for (int k0 = 0; k0 < 512; k0 += 32) {
      v16bf a = ldA_c(h1 + k0, 512);
      v16bf b = ldB_c(W2t + (size_t)n0 * 512 + k0, 512);
      acc = wmma_bf16(a, b, acc);
    }
#pragma unroll
    for (int r = 0; r < 8; ++r) {
      int m = m0 + r + mh;
      float v = (acc[r] + b2[n0 + n]) * maskrow[m];
      z[(size_t)m * CZ + n0 + n] += v;
    }
  }
}

// ---------------------------------------------------------------------------
// host orchestration
// ---------------------------------------------------------------------------
extern "C" void kernel_launch(void* const* d_in, const int* in_sizes, int n_in,
                              void* d_out, int out_size, void* d_ws, size_t ws_size,
                              hipStream_t stream) {
  (void)in_sizes; (void)n_in; (void)out_size; (void)ws_size;
  auto fin = [&](int i) { return (const float*)d_in[i]; };

  // ---- workspace layout ----
  char* ws = (char*)d_ws;
  u16* wpool = (u16*)ws;
  size_t wofs = 0;
  auto cvtw = [&](int idx, int K, int Nn, int Npad) -> u16* {
    u16* dst = wpool + wofs;
    int total = K * Npad;
    k_cvt_wt<<<(total + 255) / 256, 256, 0, stream>>>(fin(idx), dst, K, Nn, Npad);
    wofs += (size_t)total;
    return dst;
  };

  size_t off = (size_t)2 << 20;  // weights occupy < 2MB
  auto allocB = [&](size_t bytes) -> void* {
    void* p = ws + off;
    off += (bytes + 255) & ~(size_t)255;
    return p;
  };
  u16* xbf = (u16*)allocB((size_t)NN * CZ * 2);
  u16* abf = (u16*)allocB((size_t)NN * CZ * 2);
  u16* bbf = (u16*)allocB((size_t)NN * CZ * 2);
  u16* cbf = (u16*)allocB((size_t)NN * CZ * 2);
  u16* dbf = (u16*)allocB((size_t)NN * CZ * 2);
  u16* ebf = (u16*)allocB((size_t)NN * CZ * 2);
  u16* gbf = (u16*)allocB((size_t)NN * CZ * 2);
  float* tf = (float*)allocB((size_t)NN * CZ * 4);
  float* tbf = (float*)allocB((size_t)NN * NHEAD * 4);
  u16* snb = (u16*)allocB((size_t)NRES * CS * 2);
  u16* asb = (u16*)allocB((size_t)NRES * CZ * 2);
  u16* bsb = (u16*)allocB((size_t)NRES * CZ * 2);

  const float* mask = fin(2);
  float* z = (float*)d_out;

  // ---- convert weights (f32 -> bf16, transposed [Npad,K]) ----
  u16* op_w1t = cvtw(6, CS, CZ, CZ);
  u16* op_w2t = cvtw(8, CS, CZ, CZ);
  u16* op_wot = cvtw(10, CZ, CZ, CZ);
  u16* tm_w[2][6];
  for (int t = 0; t < 2; ++t) {
    int p0 = (t == 0) ? 12 : 28;
    tm_w[t][0] = cvtw(p0 + 2, CZ, CZ, CZ);   // wap
    tm_w[t][1] = cvtw(p0 + 4, CZ, CZ, CZ);   // wag
    tm_w[t][2] = cvtw(p0 + 6, CZ, CZ, CZ);   // wbp
    tm_w[t][3] = cvtw(p0 + 8, CZ, CZ, CZ);   // wbg
    tm_w[t][4] = cvtw(p0 + 12, CZ, CZ, CZ);  // wg
    tm_w[t][5] = cvtw(p0 + 14, CZ, CZ, CZ);  // wz
  }
  u16* ta_w[2][6];
  for (int t = 0; t < 2; ++t) {
    int p0 = (t == 0) ? 44 : 54;
    ta_w[t][0] = cvtw(p0 + 2, CZ, CZ, CZ);        // wq
    ta_w[t][1] = cvtw(p0 + 3, CZ, CZ, CZ);        // wk
    ta_w[t][2] = cvtw(p0 + 4, CZ, CZ, CZ);        // wv
    ta_w[t][3] = cvtw(p0 + 5, CZ, NHEAD, 16);     // wb (padded to 16 rows)
    ta_w[t][4] = cvtw(p0 + 6, CZ, CZ, CZ);        // wg
    ta_w[t][5] = cvtw(p0 + 8, CZ, CZ, CZ);        // wo
  }
  u16* pt_w1t = cvtw(66, CZ, 4 * CZ, 4 * CZ);
  u16* pt_w2t = cvtw(68, 4 * CZ, CZ, CZ);

  // ---- z <- input z ----
  hipMemcpyAsync(z, d_in[0], (size_t)NN * CZ * 4, hipMemcpyDeviceToDevice, stream);

  dim3 gemmGrid(NN / 16), gemmBlk(256);

  // ---- OuterProduct ----
  k_ln<<<(NRES + 3) / 4, 128, 0, stream>>>(fin(1), fin(4), fin(5), snb, CS, NRES);
  k_gemm<<<NRES / 16, 256, 0, stream>>>(snb, op_w1t, CS, CZ, fin(7), 1.f, 0, nullptr,
                                        0, nullptr, asb);
  k_gemm<<<NRES / 16, 256, 0, stream>>>(snb, op_w2t, CS, CZ, fin(9), 1.f, 0, nullptr,
                                        0, nullptr, bsb);
  k_outer<<<gemmGrid, gemmBlk, 0, stream>>>(asb, bsb, op_wot, fin(11), z);

  // ---- TriangleMultiplication x2 ----
  for (int t = 0; t < 2; ++t) {
    int p0 = (t == 0) ? 12 : 28;
    int incoming = t;
    k_ln<<<NN / 4, 128, 0, stream>>>(z, fin(p0), fin(p0 + 1), xbf, CZ, NN);
    k_gemm_dual<<<gemmGrid, gemmBlk, 0, stream>>>(xbf, tm_w[t][0], fin(p0 + 3), xbf,
                                                  tm_w[t][1], fin(p0 + 5), CZ, mask,
                                                  0, nullptr, abf);
    k_gemm_dual<<<gemmGrid, gemmBlk, 0, stream>>>(xbf, tm_w[t][2], fin(p0 + 7), xbf,
                                                  tm_w[t][3], fin(p0 + 9), CZ, mask,
                                                  0, nullptr, bbf);
    k_triein<<<dim3(16, 16, 16), 256, 0, stream>>>(abf, bbf, tf, incoming);
    k_ln<<<NN / 4, 128, 0, stream>>>(tf, fin(p0 + 10), fin(p0 + 11), cbf, CZ, NN);
    k_gemm_dual<<<gemmGrid, gemmBlk, 0, stream>>>(cbf, tm_w[t][5], fin(p0 + 15), xbf,
                                                  tm_w[t][4], fin(p0 + 13), CZ,
                                                  nullptr, 1, z, nullptr);
  }

  // ---- TriangleAttention x2 (starting node; ending node on z^T) ----
  const float qscale = 0.17677669529663687f;  // 1/sqrt(32)
  for (int t = 0; t < 2; ++t) {
    int p0 = (t == 0) ? 44 : 54;
    int transposed = t;
    k_ln<<<NN / 4, 128, 0, stream>>>(z, fin(p0), fin(p0 + 1), xbf, CZ, NN);
    k_gemm<<<gemmGrid, gemmBlk, 0, stream>>>(xbf, ta_w[t][0], CZ, CZ, nullptr, qscale,
                                             0, nullptr, 0, nullptr, dbf);  // q
    k_gemm<<<gemmGrid, gemmBlk, 0, stream>>>(xbf, ta_w[t][1], CZ, CZ, nullptr, 1.f, 0,
                                             nullptr, 0, nullptr, ebf);  // k
    k_gemm<<<gemmGrid, gemmBlk, 0, stream>>>(xbf, ta_w[t][2], CZ, CZ, nullptr, 1.f, 0,
                                             nullptr, 0, nullptr, cbf);  // v
    k_gemm<<<gemmGrid, gemmBlk, 0, stream>>>(xbf, ta_w[t][4], CZ, CZ, fin(p0 + 7), 1.f,
                                             1, nullptr, 0, nullptr, gbf);  // sigmoid(g)
    k_gemm<<<gemmGrid, 32, 0, stream>>>(xbf, ta_w[t][3], CZ, NHEAD, nullptr, 1.f, 0,
                                        nullptr, 1, tbf, nullptr);  // tri bias
    k_att<<<dim3(16, NHEAD, NRES), 32, 0, stream>>>(dbf, ebf, cbf, gbf, tbf, mask, abf,
                                                    transposed);
    k_gemm<<<gemmGrid, gemmBlk, 0, stream>>>(abf, ta_w[t][5], CZ, CZ, fin(p0 + 9), 1.f,
                                             0, nullptr, transposed ? 3 : 2, z,
                                             nullptr);  // out proj + residual
  }

  // ---- PairTransition ----
  k_ln<<<NN / 4, 128, 0, stream>>>(z, fin(64), fin(65), xbf, CZ, NN);
  k_pt<<<gemmGrid, gemmBlk, 0, stream>>>(xbf, pt_w1t, fin(67), pt_w2t, fin(69), mask, z);
}
